// LiquidNeuralNetwork_85177791414832
// MI455X (gfx1250) — matrix operand order repaired and checked by reading, not compiled, not run on hardware
//
#include <hip/hip_runtime.h>
#include <math.h>

// ---------------------------------------------------------------------------
// Liquid NN (ODE-RNN, RK4) for MI455X / gfx1250, wave32 + WMMA.
//
// Orientation: state kept TRANSPOSED, hT[hid, batch]:
//   k = (W_hh/tau) @ tanh(hT) + inpT - hT * (1/tau)
// Recurrent GEMM = A(static weights, registers) x B(tanh of state).
//
// Transpose-free trick: WMMA reduces over K, so the K-slot <-> hidden-row
// mapping is a free permutation as long as A and B agree. We pick the
// permutation under which each lane's C-layout accumulator VGPRs ARE the
// B-fragment elements (lane = batch column in both layouts):
//   slot s in [0,16):  row = 32*kh + (s>>3)*16 + (s&7)        (lanes 0-15)
//   slot s in [16,32): row = 32*kh + ((s-16)>>3)*16 + 8 + ((s-16)&7)
// Per eval: 32 v_tanh_f32 + 16 v_cvt_pk_f16_f32 + 8 WMMA (C seeded with inpT)
// + 32 v_fma (pre-negated 1/tau). NO LDS / barriers / shuffles in the chain.
//
// tau folded into A fragments; bias+b_comb ride a constant-1 K-slot of the x
// fragment; b_out rides the output WMMA's C seed.
// ---------------------------------------------------------------------------

typedef __attribute__((ext_vector_type(16))) _Float16 v16h;
typedef __attribute__((ext_vector_type(8)))  _Float16 v8h;
typedef __attribute__((ext_vector_type(8)))  float    v8f;
typedef __attribute__((ext_vector_type(4)))  float    v4f;

#define HID    64
#define INDIM  16

static __device__ __forceinline__ float fast_tanh(float x) {
#if __has_builtin(__builtin_amdgcn_tanhf)
  return __builtin_amdgcn_tanhf(x);          // v_tanh_f32
#elif __has_builtin(__builtin_amdgcn_tanh_f32)
  return __builtin_amdgcn_tanh_f32(x);
#else
  return tanhf(x);
#endif
}

static __device__ __forceinline__ v16h cat8(v8h lo, v8h hi) {
  return __builtin_shufflevector(lo, hi, 0,1,2,3,4,5,6,7,8,9,10,11,12,13,14,15);
}

static __device__ __forceinline__ v8f wmma16(v16h a, v16h b, v8f c) {
  // D = A(16x32 f16) * B(32x16 f16) + C(16x16 f32)
  return __builtin_amdgcn_wmma_f32_16x16x32_f16(false, a, false, b, (short)0, c, false, false);
}

// K-slot (within the 64-wide K space) -> hidden row, per the permutation above.
static __device__ __forceinline__ int rowmap(int k) {
  const int base = (k >> 5) << 5;
  const int s = k & 31;
  return (s < 16) ? (base + ((s >> 3) << 4) + (s & 7))
                  : (base + (((s - 16) >> 3) << 4) + 8 + ((s - 16) & 7));
}

// K numbering chosen here (A and B alike): fragment element e of lane l covers K-slot:
static __device__ __forceinline__ int slot_of(int l, int e) {
  return (l < 16) ? e : e + 16;
}

// B fragments of tanh(state): pure VALU, identical code on both lane halves.
static __device__ __forceinline__ void build_b(const v8f (&st)[4], v16h &B0, v16h &B1) {
  #pragma unroll
  for (int e = 0; e < 8; ++e) {
    B0[e]     = (_Float16)fast_tanh(st[0][e]);
    B0[8 + e] = (_Float16)fast_tanh(st[1][e]);
    B1[e]     = (_Float16)fast_tanh(st[2][e]);
    B1[8 + e] = (_Float16)fast_tanh(st[3][e]);
  }
}

// kout := (W_hh*itau) @ tanh(hin) + inp - hin*itau   (kout may alias hin)
static __device__ __forceinline__ void ode_eval(
    const v8f (&hin)[4], v8f (&kout)[4], const v8f (&inp)[4],
    const v16h (&Aw)[4][2], const v8f (&nitau)[4])
{
  v16h B0, B1;
  build_b(hin, B0, B1);
  #pragma unroll
  for (int mt = 0; mt < 4; ++mt) {
    v8f c = wmma16(Aw[mt][0], B0, inp[mt]);   // C seeded with inpT: free add
    c = wmma16(Aw[mt][1], B1, c);
    #pragma unroll
    for (int v = 0; v < 8; ++v)
      kout[mt][v] = __builtin_fmaf(hin[mt][v], nitau[mt][v], c[v]);
  }
}

__global__ void __launch_bounds__(32)
lnn_rk4_kernel(const float* __restrict__ x,
               const float* __restrict__ W_in, const float* __restrict__ b_in,
               const float* __restrict__ W_hh, const float* __restrict__ W_ih,
               const float* __restrict__ bias, const float* __restrict__ tau,
               const float* __restrict__ W_out, const float* __restrict__ b_out,
               float* __restrict__ out, int S)
{
  __shared__ _Float16 ldsAwc[4 * 32 * 16];   // 4 KB: input-proj A fragments
  __shared__ _Float16 ldsAwo[2 * 32 * 16];   // 2 KB: output A fragments

  const int l   = threadIdx.x;               // 0..31 (wave32)
  const int lan = l & 15;
  const int hi8 = (l < 16) ? 0 : 8;
  const int m0  = blockIdx.x * 16;           // batch-column base of this wave

  // ---- A_whh fragments: W_hh * (1/tau) with the K-permutation baked in ----
  v16h Aw[4][2];
  #pragma unroll
  for (int mt = 0; mt < 4; ++mt) {
    const int m_out = mt * 16 + lan;
    const float sc = 1.f / tau[m_out];
    #pragma unroll
    for (int kh = 0; kh < 2; ++kh) {
      v16h a;
      #pragma unroll
      for (int e = 0; e < 16; ++e) {
        const int rin = rowmap(kh * 32 + slot_of(l, e));
        a[e] = (_Float16)(W_hh[m_out * HID + rin] * sc);
      }
      Aw[mt][kh] = a;
    }
  }

  // ---- negated 1/tau per (tile, vgpr): row = mt*16 + hi8 + v ---------------
  v8f nitau[4];
  #pragma unroll
  for (int mt = 0; mt < 4; ++mt)
    #pragma unroll
    for (int v = 0; v < 8; ++v) nitau[mt][v] = -1.f / tau[mt * 16 + hi8 + v];

  // ---- A_wc = (W_ih@W_in)*itau, with (W_ih@b_in + bias)*itau in K-slot 16 --
  #pragma unroll
  for (int mt = 0; mt < 4; ++mt) {
    const int m_out = mt * 16 + lan;
    const float sc = 1.f / tau[m_out];
    const float* wr = W_ih + m_out * HID;
    float wc[INDIM];
    #pragma unroll
    for (int j = 0; j < INDIM; ++j) wc[j] = 0.f;
    float bc = bias[m_out];
    for (int k = 0; k < HID; ++k) {
      const float w = wr[k];
      bc += w * b_in[k];
      #pragma unroll
      for (int j = 0; j < INDIM; ++j) wc[j] += w * W_in[k * INDIM + j];
    }
    v16h a;
    #pragma unroll
    for (int e = 0; e < 16; ++e) {
      const int sl = slot_of(l, e);
      a[e] = (sl < INDIM) ? (_Float16)(wc[sl] * sc)
           : (sl == INDIM) ? (_Float16)(bc * sc) : (_Float16)0.f;
    }
    _Float16* ap = ldsAwc + (mt * 32 + l) * 16;
    *(v8h*)ap       = __builtin_shufflevector(a, a, 0,1,2,3,4,5,6,7);
    *(v8h*)(ap + 8) = __builtin_shufflevector(a, a, 8,9,10,11,12,13,14,15);
  }

  // ---- A_wout: W_out in matrix row 0 (lanes with lan==0), permuted K -------
  #pragma unroll
  for (int kh = 0; kh < 2; ++kh) {
    v16h a;
    #pragma unroll
    for (int e = 0; e < 16; ++e) {
      const int rin = rowmap(kh * 32 + slot_of(l, e));
      a[e] = (lan == 0) ? (_Float16)W_out[rin] : (_Float16)0.f;
    }
    _Float16* ap = ldsAwo + (kh * 32 + l) * 16;
    *(v8h*)ap       = __builtin_shufflevector(a, a, 0,1,2,3,4,5,6,7);
    *(v8h*)(ap + 8) = __builtin_shufflevector(a, a, 8,9,10,11,12,13,14,15);
  }
  __syncthreads();

  const float b_out0 = b_out[0];

  // ---- state hT (C-layout: lane = batch col, vgpr = hidden row) ------------
  v8f h[4];
  #pragma unroll
  for (int mt = 0; mt < 4; ++mt)
    #pragma unroll
    for (int v = 0; v < 8; ++v) h[mt][v] = 0.f;

  const float stepdt = 1.f / (float)(S - 1);   // linspace(0,1,S); dts[0] = 0

  for (int s = 0; s < S; ++s) {
    const float dt   = (s == 0) ? 0.f : stepdt;
    const float hsub = dt * 0.25f;             // dt / N_SUB

    // ---- x^T B-fragment: lanes<16 carry x[n][0..15]; lanes>=16 carry the
    //      constant-1 slot (slot 16) that injects the bias column.
    const float* xp = x + ((size_t)(m0 + lan) * S + s) * INDIM;
    v16h Bx;
    if (l < 16) {
      v4f a0 = *(const v4f*)xp;
      v4f a1 = *(const v4f*)(xp + 4);
      v4f a2 = *(const v4f*)(xp + 8);
      v4f a3 = *(const v4f*)(xp + 12);
      #pragma unroll
      for (int j = 0; j < 4; ++j) {
        Bx[j] = (_Float16)a0[j]; Bx[4 + j]  = (_Float16)a1[j];
        Bx[8 + j] = (_Float16)a2[j]; Bx[12 + j] = (_Float16)a3[j];
      }
    } else {
      #pragma unroll
      for (int e = 0; e < 16; ++e) Bx[e] = (_Float16)0.f;
      Bx[0] = (_Float16)1.f;                  // K-slot 16 == 1.0
    }
    if (s + 1 < S) __builtin_prefetch(xp + INDIM, 0, 1);   // global_prefetch_b8

    // ---- inpT = (W_comb x_t + b_comb + bias) * itau, via 4 WMMAs -----------
    v8f inp[4];
    #pragma unroll
    for (int mt = 0; mt < 4; ++mt) {
      const _Float16* ap = ldsAwc + (mt * 32 + l) * 16;
      v16h a = cat8(*(const v8h*)ap, *(const v8h*)(ap + 8));
      v8f c;
      #pragma unroll
      for (int v = 0; v < 8; ++v) c[v] = 0.f;
      inp[mt] = wmma16(a, Bx, c);
    }

    // ---- RK4 with N_SUB = 4 substeps (uniform code, no k1 special case) ----
    for (int sub = 0; sub < 4; ++sub) {
      v8f he[4], acc[4];
      ode_eval(h, he, inp, Aw, nitau);                           // he = k1
      #pragma unroll
      for (int mt = 0; mt < 4; ++mt) { acc[mt] = he[mt]; he[mt] = h[mt] + he[mt] * (0.5f * hsub); }
      ode_eval(he, he, inp, Aw, nitau);                          // he = k2
      #pragma unroll
      for (int mt = 0; mt < 4; ++mt) { acc[mt] += he[mt] * 2.f; he[mt] = h[mt] + he[mt] * (0.5f * hsub); }
      ode_eval(he, he, inp, Aw, nitau);                          // he = k3
      #pragma unroll
      for (int mt = 0; mt < 4; ++mt) { acc[mt] += he[mt] * 2.f; he[mt] = h[mt] + he[mt] * hsub; }
      ode_eval(he, he, inp, Aw, nitau);                          // he = k4
      #pragma unroll
      for (int mt = 0; mt < 4; ++mt) { acc[mt] += he[mt]; h[mt] += acc[mt] * (hsub * (1.f / 6.f)); }
    }

    // ---- out[b,s,0] = tanh(h)@W_out^T + b_out, as 2 WMMAs ------------------
    v16h B0, B1;
    build_b(h, B0, B1);
    const _Float16* o0 = ldsAwo + (0 * 32 + l) * 16;
    const _Float16* o1 = ldsAwo + (1 * 32 + l) * 16;
    v8f oc;
    #pragma unroll
    for (int v = 0; v < 8; ++v) oc[v] = 0.f;
    oc[0] = b_out0;                                       // b_out rides the C seed
    oc = wmma16(cat8(*(const v8h*)o0, *(const v8h*)(o0 + 8)), B0, oc);
    oc = wmma16(cat8(*(const v8h*)o1, *(const v8h*)(o1 + 8)), B1, oc);
    if (l < 16)
      out[(size_t)(m0 + lan) * S + s] = oc[0];            // D row 0, lanes 0-15
  }
}

extern "C" void kernel_launch(void* const* d_in, const int* in_sizes, int n_in,
                              void* d_out, int out_size, void* d_ws, size_t ws_size,
                              hipStream_t stream) {
  (void)n_in; (void)d_ws; (void)ws_size; (void)in_sizes;
  const float* x     = (const float*)d_in[0];
  const float* W_in  = (const float*)d_in[1];
  const float* b_in  = (const float*)d_in[2];
  const float* W_hh  = (const float*)d_in[3];
  const float* W_ih  = (const float*)d_in[4];
  const float* bias  = (const float*)d_in[5];
  const float* tau   = (const float*)d_in[6];
  const float* W_out = (const float*)d_in[7];
  const float* b_out = (const float*)d_in[8];
  float* out = (float*)d_out;

  const int S = 1024;                       // per reference setup
  const int B = out_size / S;               // 512

  // 32 single-wave workgroups; each wave owns a 16-column batch tile.
  lnn_rk4_kernel<<<B / 16, 32, 0, stream>>>(x, W_in, b_in, W_hh, W_ih,
                                            bias, tau, W_out, b_out, out, S);
}
